// GENEO_76742475645631
// MI455X (gfx1250) — compile-verified, matrix operand order checked by reading, not run on hardware
//
#include <hip/hip_runtime.h>
#include <math.h>

// ---------------------------------------------------------------------------
// GENEO pipeline for MI455X (gfx1250, wave32).
//   geneo1 : SAD maps; the reduction over the 729 patch elements runs on the
//            matrix pipe with V_WMMA_F32_16X16X32_F16 (all-ones A => per-
//            column sums, f32 accumulation), co-executing with the VALU diff
//            stream. Diffs are produced in f32 (v_pk_add neg pairs),
//            truncated to f16 (v_cvt_pk), |.| applied as a packed sign-clear
//            AND in the f16 domain (identical to abs-before-truncate; avoids
//            the gfx1250 ISel crash on vector f32 fabs). LDS offset table
//            kills per-element /27,%27 math; zero-padded pattern/offset/image
//            tables make the K-loop completely branch-free (736 = 23*32).
//   geneo3 : bilinear-shift gather weighted sum over k.
//   final  : softmax(10*T) . T  -> scalar.
// ---------------------------------------------------------------------------

typedef __attribute__((ext_vector_type(16))) _Float16     v16h;
typedef __attribute__((ext_vector_type(8)))  float        v8f;
typedef __attribute__((ext_vector_type(8)))  unsigned int v8u;

#define HIMG   28
#define NPIX   784          // 28*28 = 49 tiles of 16
#define HPAD   54           // 28 + 2*13
#define HP     27
#define PATLEN 729          // 27*27
#define PPAD   736          // 23 full 32-chunks, zero-padded tail
#define XPIMG  2916         // 54*54
#define XPTOT  4416         // 2916 + zero region covering ibase(max)=1485
#define KPAT   256

// ---------------------------------------------------------------------------
// Kernel 1: one block per pattern k. 256 threads = 8 waves.
// Each wave owns 16-pixel tiles (49 tiles, round-robin).
// Inner loop: 32 patch elements -> one v_wmma_f32_16x16x32_f16.
//   lane = pixel column (lanes 0..15 / 16..31 supply the two 16-K halves);
//   all-ones A makes every row of D the per-pixel column sum, independent of
//   the K-slot permutation within a column. Padded slots contribute 0-0=0.
// ---------------------------------------------------------------------------
__global__ __launch_bounds__(256) void geneo1_wmma_kernel(
    const float* __restrict__ x,          // (28*28)
    const float* __restrict__ patterns,   // (256*729)
    float* __restrict__ F)                // out: (256*784)
{
    __shared__ float s_xp[XPTOT];         // padded image + zero tail (17.7 KB)
    __shared__ float s_pat[PPAD];         // pattern, zero-padded (2.9 KB)
    __shared__ int   s_off[PPAD];         // p -> dy*54+dx; pad -> zero region

    const int k   = blockIdx.x;
    const int tid = threadIdx.x;

    // Stage padded image into LDS; indices >= 2916 fall out of the 28x28
    // range automatically and get 0 (this is the zero region the padded
    // offsets point into).
    for (int idx = tid; idx < XPTOT; idx += 256) {
        const int r  = idx / HPAD;
        const int c  = idx - r * HPAD;
        const int rr = r - 13;
        const int cc = c - 13;
        float v = 0.0f;
        if (rr >= 0 && rr < HIMG && cc >= 0 && cc < HIMG)
            v = x[rr * HIMG + cc];
        s_xp[idx] = v;
    }
    // Stage this block's pattern + window-offset table (both zero-padded).
    for (int idx = tid; idx < PPAD; idx += 256) {
        if (idx < PATLEN) {
            s_pat[idx] = patterns[k * PATLEN + idx];
            const int dy = idx / HP;
            s_off[idx] = dy * HPAD + (idx - dy * HP);
        } else {
            s_pat[idx] = 0.0f;
            s_off[idx] = XPIMG;   // ibase + XPIMG <= 4401 < XPTOT, always 0
        }
    }
    __syncthreads();

    const int lane = tid & 31;
    const int wave = tid >> 5;          // 0..7
    const int col  = lane & 15;         // pixel column within the 16-tile
    const int pb   = (lane >> 4) * 16;  // which 16 of the 32 K-values

    v16h ones;
#pragma unroll
    for (int e = 0; e < 16; ++e) ones[e] = (_Float16)1.0f;

    for (int t = wave; t < NPIX / 16; t += 8) {
        const int q = t * 16 + col;          // pixel linear index, 0..783
        const int i = q / HIMG;
        const int j = q - i * HIMG;
        const int ibase = i * HPAD + j;      // top-left of this pixel's window

        v8f acc = {};
        for (int cb = 0; cb < PPAD; cb += 32) {   // 23 branch-free chunks
            v16h b;
#pragma unroll
            for (int e = 0; e < 16; ++e) {
                const int p = cb + pb + e;
                const float d = s_pat[p] - s_xp[ibase + s_off[p]];
                b[e] = (_Float16)d;          // sign-symmetric rounding
            }
            // |.| on all 16 halves: packed sign-clear (8x v_and_b32).
            v8u ub = __builtin_bit_cast(v8u, b);
#pragma unroll
            for (int w = 0; w < 8; ++w) ub[w] &= 0x7fff7fffu;
            b = __builtin_bit_cast(v16h, ub);

            // D = ones(16x32) * B(32x16) + C : every row = column sums.
            acc = __builtin_amdgcn_wmma_f32_16x16x32_f16(
                false, ones, false, b, (short)0, acc, false, false);
        }

        // acc[0] at lane L holds D[0, L&15] (rows identical) = sum for pixel q.
        if (lane < 16)
            F[k * NPIX + q] = 1.0f - acc[0] * (1.0f / 729.0f);
    }
}

// ---------------------------------------------------------------------------
// Kernel 2: geneo3. Grid = 49 blocks x 256 threads; block handles 16 pixels.
// Threads split as (pixel = tid&15, k-group = tid>>4); 16 groups stride over
// the 255 vectors, partials reduced through LDS.
// ---------------------------------------------------------------------------
__global__ __launch_bounds__(256) void geneo3_kernel(
    const float* __restrict__ F,          // (256*784)
    const float* __restrict__ vectors,    // (255*2)
    float* __restrict__ T)                // out: (784)
{
    __shared__ int   s_vx[255], s_vy[255];
    __shared__ float s_w0[255], s_w1[255], s_w2[255], s_w3[255];
    __shared__ float s_part[256];

    const int tid = threadIdx.x;

    if (tid < 255) {
        const float vxf = vectors[2 * tid + 0];
        const float vyf = vectors[2 * tid + 1];
        const float fvx = floorf(vxf);
        const float fvy = floorf(vyf);
        const float px  = vxf - fvx;
        const float py  = vyf - fvy;
        s_vx[tid] = (int)fvx;
        s_vy[tid] = (int)fvy;
        s_w0[tid] = px * py;                    // corner (vx,   vy  )
        s_w1[tid] = px * (1.0f - py);           // corner (vx,   vy+1)
        s_w2[tid] = (1.0f - px) * py;           // corner (vx+1, vy  )
        s_w3[tid] = (1.0f - px) * (1.0f - py);  // corner (vx+1, vy+1)
    }
    __syncthreads();

    const int pl  = tid & 15;
    const int grp = tid >> 4;                  // 0..15
    const int q   = blockIdx.x * 16 + pl;      // pixel 0..783
    const int a   = q / HIMG;
    const int bcl = q - a * HIMG;

    float acc = 0.0f;
    for (int k = grp; k < 255; k += 16) {
        const int vx = s_vx[k];
        const int vy = s_vy[k];
        const int ii0 = min(max(a   - vx,     0), HIMG - 1);
        const int ii1 = min(max(a   - vx - 1, 0), HIMG - 1);
        const int jj0 = min(max(bcl - vy,     0), HIMG - 1);
        const int jj1 = min(max(bcl - vy - 1, 0), HIMG - 1);
        const float* Fk = F + k * NPIX;
        acc += s_w0[k] * Fk[ii0 * HIMG + jj0]
             + s_w1[k] * Fk[ii0 * HIMG + jj1]
             + s_w2[k] * Fk[ii1 * HIMG + jj0]
             + s_w3[k] * Fk[ii1 * HIMG + jj1];
    }
    s_part[tid] = acc;
    __syncthreads();

    for (int s = 8; s >= 1; s >>= 1) {
        if (grp < s)
            s_part[grp * 16 + pl] += s_part[(grp + s) * 16 + pl];
        __syncthreads();
    }
    if (grp == 0) {
        const float tot = s_part[pl] + F[255 * NPIX + q];   // += F[-1,0]
        T[q] = tot * (1.0f / (float)KPAT);
    }
}

// ---------------------------------------------------------------------------
// Kernel 3: scalar = softmax(10*T) . T  (max-stabilized, fixed-order reduce).
// ---------------------------------------------------------------------------
__global__ __launch_bounds__(256) void softdot_kernel(
    const float* __restrict__ T, float* __restrict__ out)
{
    __shared__ float red[256];
    const int tid = threadIdx.x;

    float lm = -3.402823466e38f;
    for (int i = tid; i < NPIX; i += 256)
        lm = fmaxf(lm, 10.0f * T[i]);
    red[tid] = lm;
    __syncthreads();
    for (int s = 128; s >= 1; s >>= 1) {
        if (tid < s) red[tid] = fmaxf(red[tid], red[tid + s]);
        __syncthreads();
    }
    const float m = red[0];
    __syncthreads();

    float s1 = 0.0f, s2 = 0.0f;
    for (int i = tid; i < NPIX; i += 256) {
        const float f = T[i];
        const float e = expf(10.0f * f - m);
        s1 += e;
        s2 += e * f;
    }
    red[tid] = s1;
    __syncthreads();
    for (int s = 128; s >= 1; s >>= 1) {
        if (tid < s) red[tid] += red[tid + s];
        __syncthreads();
    }
    const float S1 = red[0];
    __syncthreads();
    red[tid] = s2;
    __syncthreads();
    for (int s = 128; s >= 1; s >>= 1) {
        if (tid < s) red[tid] += red[tid + s];
        __syncthreads();
    }
    if (tid == 0)
        out[0] = red[0] / S1;
}

// ---------------------------------------------------------------------------
extern "C" void kernel_launch(void* const* d_in, const int* in_sizes, int n_in,
                              void* d_out, int out_size, void* d_ws, size_t ws_size,
                              hipStream_t stream)
{
    const float* x        = (const float*)d_in[0];   // (1,1,28,28)
    const float* patterns = (const float*)d_in[1];   // (256,1,27,27)
    const float* vectors  = (const float*)d_in[2];   // (255,2)

    float* F = (float*)d_ws;                // 256*784 floats (784 KB)
    float* T = F + KPAT * NPIX;             // 784 floats
    float* out = (float*)d_out;

    geneo1_wmma_kernel<<<KPAT, 256, 0, stream>>>(x, patterns, F);
    geneo3_kernel<<<NPIX / 16, 256, 0, stream>>>(F, vectors, T);
    softdot_kernel<<<1, 256, 0, stream>>>(T, out);
}
